// KANConv_41068477284778
// MI455X (gfx1250) — matrix operand order repaired and checked by reading, not compiled, not run on hardware
//
#include <hip/hip_runtime.h>

// ---------------------------------------------------------------------------
// KAN conv as implicit GEMM, bf16 WMMA, f32 accumulate.
//   out[b,cout,h,w] = sum_{kh,kw,cin} Wspl[cout,cin,kh,kw] * x[b,cin,h+kh-1,w+kw-1]
//                   + sum_{kh,kw,cin} Wbas[cout,cin,kh,kw] * silu(x)[...]
// K folded as k = branch*64 + cin (128); 9 taps = 9 shifted GEMMs into the
// same accumulators. HT=4 output rows per block share one 6-row staged tile:
// x read overhead drops from 3x to 1.5x  (~232 MB total -> ~10 us @ 23.3 TB/s).
// ---------------------------------------------------------------------------

typedef __attribute__((ext_vector_type(16))) __bf16 v16bf;
typedef __attribute__((ext_vector_type(8)))  float  v8f;

#define B_    16
#define CIN_  64
#define COUT_ 128
#define HH_   128
#define WW_   128
#define NTILE 64          // W-tile per block
#define HT    4           // output rows per block
#define XR    (HT + 2)    // staged rows (with halo)
#define XC    66          // staged columns: NTILE + 2 halo
#define KS    136         // padded k-stride in LDS: 272B = 68 banks -> conflict-free,
                          // and keeps 16B alignment for ds_load_b128

union Frag { uint4 u[2]; v16bf v; };

static __device__ inline unsigned short bf16bits(float f) {
  __bf16 b = (__bf16)f;
  return __builtin_bit_cast(unsigned short, b);
}

// ---------------------------------------------------------------------------
// Prep: pack both weight branches into pre-swizzled WMMA A-fragment order:
//   apack[tap(9)][kstep(4)][mtile(8)][lane(32)][dword j(8)][half(2)]  (bf16)
// A 16x32 bf16 fragment layout (per ISA):
//   m = lane%16 ; lanes 0-15 hold K 0..7 (v0-3) & 16..23 (v4-7);
//   lanes 16-31 hold K 8..15 & 24..31; each dword packs K even(lo)/odd(hi).
// ---------------------------------------------------------------------------
__global__ __launch_bounds__(256) void kan_pack_weights(
    const float* __restrict__ spline,   // (COUT, CIN, 3, 3, 4)
    const float* __restrict__ basis,    // (COUT, CIN, 3, 3)
    unsigned short* __restrict__ apack) {
  int e = blockIdx.x * 256 + threadIdx.x;        // 0 .. 9*16384-1
  int half = e & 1;
  int j    = (e >> 1) & 7;
  int lane = (e >> 4) & 31;
  int mt   = (e >> 9) & 7;
  int s    = (e >> 12) & 3;
  int tap  = e >> 14;
  if (tap >= 9) return;

  int kloc = ((j < 4) ? 0 : 16) + ((lane >= 16) ? 8 : 0) + (j & 3) * 2 + half;
  int k = s * 32 + kloc;                  // 0..127 : branch*64 + cin
  int m = mt * 16 + (lane & 15);          // cout
  int branch = k >> 6, cin = k & 63;
  int kh = tap / 3, kw = tap - kh * 3;

  int wi = ((m * CIN_ + cin) * 3 + kh) * 3 + kw;
  float v;
  if (branch == 0) {
    const float* p = spline + wi * 4;     // sum over grid G=4
    v = p[0] + p[1] + p[2] + p[3];
  } else {
    v = basis[wi];
  }
  apack[e] = bf16bits(v);
}

// ---------------------------------------------------------------------------
// Main kernel: block = (w-tile, h-tile of 4 rows, b). Per row: GEMM tile
// M=128 (all cout) x N=64. 8 waves: wm=wave%4 -> 2 M-subtiles, wn=wave/4 ->
// 2 N-subtiles; 4 v8f accumulators per wave.
// ---------------------------------------------------------------------------
__global__ __launch_bounds__(256) void kan_conv_wmma(
    const float* __restrict__ x,                // (B, CIN, H, W)
    const unsigned short* __restrict__ apack,   // packed weights (bf16 bits)
    float* __restrict__ out) {                  // (B, COUT, H, W)
  __shared__ unsigned short xs[XR * XC * KS];   // 107,712 B (<= 320KB/WGP on CDNA5)

  const int n0  = blockIdx.x * NTILE;
  const int h0  = blockIdx.y * HT;
  const int b   = blockIdx.z;
  const int tid = threadIdx.x;

  // ---- stage (HT+2)-row halo tile: x (k=0..63) and silu(x) (k=64..127) ----
  for (int idx = tid; idx < XR * XC * CIN_; idx += 256) {
    int c   = idx % XC;           // 0..65  -> w' = n0 + c - 1
    int t   = idx / XC;
    int cin = t % CIN_;
    int r   = t / CIN_;           // 0..XR-1 -> h' = h0 + r - 1
    int hp = h0 + r - 1;
    int wp = n0 + c - 1;
    float v = 0.0f;
    if ((unsigned)hp < (unsigned)HH_ && (unsigned)wp < (unsigned)WW_)
      v = x[((b * CIN_ + cin) * HH_ + hp) * WW_ + wp];
    float sv = v / (1.0f + __expf(-v));         // silu(x); silu(0)==0 at pads
    int base = (r * XC + c) * KS;
    xs[base + cin]      = bf16bits(v);
    xs[base + 64 + cin] = bf16bits(sv);
  }
  __syncthreads();

  const int lane = tid & 31;
  const int wave = tid >> 5;
  const int wm   = wave & 3;      // m-group: covers mtiles wm*2, wm*2+1
  const int wn   = wave >> 2;     // n-group: covers cols wn*32 .. wn*32+31
  const int lrow = lane & 15;
  const int lhi  = lane >> 4;

  const uint4* ap = (const uint4*)apack;

#pragma unroll 1                  // keep the row loop rolled (LDS tile reuse)
  for (int hi = 0; hi < HT; ++hi) {
    v8f acc[2][2] = {};           // zero-initialized accumulators

    for (int tap = 0; tap < 9; ++tap) {
      const int kh = tap / 3;
      const int kw = tap - kh * 3;
      const int c0 = wn * 32 + lrow + kw;       // B-frag col, subtile 0
      const int rowbase = (hi + kh) * XC;       // staged row for this tap

      // cover L2->L0 latency of the next tap's weight stream
      if (tap < 8)
        __builtin_prefetch(&ap[(((tap + 1) * 4) * 8) * 64 + lane * 2], 0, 1);

#pragma unroll
      for (int s = 0; s < 4; ++s) {
        // A fragments: 32 contiguous bytes per lane from pre-swizzled pack
        Frag a0, a1, b0, b1;
        int abase = (((tap * 4 + s) * 8 + wm * 2) * 32 + lane) * 2;  // uint4 units
        a0.u[0] = ap[abase + 0];   a0.u[1] = ap[abase + 1];
        a1.u[0] = ap[abase + 64];  a1.u[1] = ap[abase + 65];  // next mtile (+32 lanes)

        // B fragments: two 16B k-chunks per lane from LDS (K 8h..8h+7, 16+8h..23+8h)
        int koff = s * 32 + (lhi ? 8 : 0);
        const uint4* l0 = (const uint4*)&xs[(rowbase + c0) * KS + koff];
        const uint4* l1 = (const uint4*)&xs[(rowbase + c0 + 16) * KS + koff];
        b0.u[0] = l0[0]; b0.u[1] = l0[2];
        b1.u[0] = l1[0]; b1.u[1] = l1[2];

        acc[0][0] = __builtin_amdgcn_wmma_f32_16x16x32_bf16(false, a0.v, false, b0.v,
                                                            (short)0, acc[0][0], false, false);
        acc[0][1] = __builtin_amdgcn_wmma_f32_16x16x32_bf16(false, a0.v, false, b1.v,
                                                            (short)0, acc[0][1], false, false);
        acc[1][0] = __builtin_amdgcn_wmma_f32_16x16x32_bf16(false, a1.v, false, b0.v,
                                                            (short)0, acc[1][0], false, false);
        acc[1][1] = __builtin_amdgcn_wmma_f32_16x16x32_bf16(false, a1.v, false, b1.v,
                                                            (short)0, acc[1][1], false, false);
      }
    }

    // ---- epilogue: f32 C fragment -> NCHW, non-temporal (never re-read) ----
    // C layout: VGPR i: lanes 0-15 = (M=i, N=lane), lanes 16-31 = (M=8+i, N=lane-16)
    const int h = h0 + hi;
#pragma unroll
    for (int i = 0; i < 2; ++i) {
#pragma unroll
      for (int j = 0; j < 2; ++j) {
        int mbase = (wm * 2 + i) * 16 + lhi * 8;
        int w = n0 + wn * 32 + j * 16 + lrow;
        v8f a = acc[i][j];
#pragma unroll
        for (int v = 0; v < 8; ++v) {
          int m = mbase + v;
          __builtin_nontemporal_store(a[v], &out[((b * COUT_ + m) * HH_ + h) * WW_ + w]);
        }
      }
    }
  }
}

extern "C" void kernel_launch(void* const* d_in, const int* in_sizes, int n_in,
                              void* d_out, int out_size, void* d_ws, size_t ws_size,
                              hipStream_t stream) {
  const float* x      = (const float*)d_in[0];   // (16, 64, 128, 128)
  const float* spline = (const float*)d_in[1];   // (128, 64, 3, 3, 4)
  const float* basis  = (const float*)d_in[2];   // (128, 64, 3, 3)
  float* out = (float*)d_out;                    // (16, 128, 128, 128)
  unsigned short* apack = (unsigned short*)d_ws; // 9*128*128 bf16 = 294,912 B

  kan_pack_weights<<<576, 256, 0, stream>>>(spline, basis, apack);

  dim3 grid(WW_ / NTILE, HH_ / HT, B_);          // (2, 32, 16)
  kan_conv_wmma<<<grid, 256, 0, stream>>>(x, apack, out);
}
